// Net_1022202216751
// MI455X (gfx1250) — matrix operand order repaired and checked by reading, not compiled, not run on hardware
//
#include <hip/hip_runtime.h>
#include <hip/hip_bf16.h>
#include <math.h>

// ---------------------------------------------------------------------------
// MI455X (gfx1250) implementation.
// GEMM-shaped ops (implicit-GEMM 3x3 convs, 1x1 convs, mamba projections) go
// through v_wmma_f32_16x16x32_bf16 (wave32, fp32 accum). Multi-N-tile waves
// amortize staging; global->LDS staging uses b128 vector loads on the fast
// path. Elementwise / reduction / scan parts stay fp32 VALU.
// ---------------------------------------------------------------------------

typedef __attribute__((ext_vector_type(16))) __bf16 v16bf;
typedef __attribute__((ext_vector_type(8)))  float  v8f;

union BFV { v16bf v; unsigned short s[16]; };

static __device__ inline unsigned short f2bf(float f) {
  unsigned int u = __float_as_uint(f);
  unsigned int r = (u + 0x7fffu + ((u >> 16) & 1u)) >> 16;   // RNE
  return (unsigned short)r;
}
static __device__ inline int iclamp(int v, int lo, int hi) {
  return v < lo ? lo : (v > hi ? hi : v);
}
static __device__ inline float lrelu(float v) { return v > 0.f ? v : 0.2f * v; }
static __device__ inline float siluf(float v) { return v / (1.f + expf(-v)); }

#define ACT_NONE  0
#define ACT_LRELU 1

// Stage a 16x32 fp32 tile (rows rowBase..+15 of src[nRows x ld], cols kk..+31)
// into LDS as bf16. Fast path: one lane = one half-row = 4 x b128 loads.
static __device__ inline void stage_tile(const float* __restrict__ src, int rowBase, int nRows,
                                         int ld, int kk, int K,
                                         unsigned short (*sT)[32], int lane) {
  const int r  = lane >> 1;
  const int c0 = (lane & 1) * 16;
  const bool fast = ((ld & 3) == 0) && (rowBase + 15 < nRows) && (kk + 32 <= K);
  if (fast) {
    const float4* p = (const float4*)(src + (size_t)(rowBase + r) * ld + kk + c0);
#pragma unroll
    for (int j = 0; j < 4; ++j) {
      float4 v = p[j];
      int c = c0 + j * 4;
      sT[r][c]     = f2bf(v.x);
      sT[r][c + 1] = f2bf(v.y);
      sT[r][c + 2] = f2bf(v.z);
      sT[r][c + 3] = f2bf(v.w);
    }
  } else {
#pragma unroll
    for (int j = 0; j < 16; ++j) {
      int c = c0 + j;
      int rr = rowBase + r, k = kk + c;
      float v = (rr < nRows && k < K) ? src[(size_t)rr * ld + k] : 0.f;
      sT[r][c] = f2bf(v);
    }
  }
}

// Fragment builders per the CDNA5 16-bit A(16x32) / B(32x16) lane layouts.
static __device__ inline v16bf frag_a(const unsigned short (*sA)[32], int lane) {
  BFV a;
  const int row = lane & 15;
  const int kb = (lane < 16) ? 0 : 8;
#pragma unroll
  for (int j = 0; j < 8; ++j) { a.s[j] = sA[row][kb + j]; a.s[j + 8] = sA[row][16 + kb + j]; }
  return a.v;
}
static __device__ inline v16bf frag_b(const unsigned short (*sB)[32], int lane) {
  BFV b;
  const int col = lane & 15;
  const int kh = (lane < 16) ? 0 : 8;
#pragma unroll
  for (int j = 0; j < 8; ++j) { b.s[j] = sB[col][kh + j]; b.s[j + 8] = sB[col][16 + kh + j]; }
  return b.v;
}

// ---------- WMMA GEMM: C[M,N] = act(A[M,K] * W[N,K]^T + b), NT n-tiles/wave --
template <int NT>
__global__ void k_gemm_wmma(const float* __restrict__ A, const float* __restrict__ Wt,
                            const float* __restrict__ bias, float* __restrict__ C,
                            int M, int K, int N, int act) {
  __shared__ unsigned short sA[16][32];
  __shared__ unsigned short sB[NT][16][32];
  const int lane = threadIdx.x;            // wave32, one M-tile x NT N-tiles
  const int mBase = blockIdx.x * 16;
  const int nBase = blockIdx.y * (16 * NT);
  v8f acc[NT];
#pragma unroll
  for (int t = 0; t < NT; ++t)
    for (int i = 0; i < 8; ++i) acc[t][i] = 0.f;

  for (int kk = 0; kk < K; kk += 32) {
    if (kk + 32 < K) __builtin_prefetch(&A[(size_t)mBase * K + kk + 32], 0, 0);
    stage_tile(A, mBase, M, K, kk, K, sA, lane);
#pragma unroll
    for (int t = 0; t < NT; ++t)
      stage_tile(Wt, nBase + 16 * t, N, K, kk, K, sB[t], lane);
    __syncthreads();
    v16bf a = frag_a(sA, lane);
#pragma unroll
    for (int t = 0; t < NT; ++t) {
      v16bf b = frag_b(sB[t], lane);
      acc[t] = __builtin_amdgcn_wmma_f32_16x16x32_bf16(false, a, false, b,
                                                       (short)0, acc[t], false, false);
    }
    __syncthreads();
  }
#pragma unroll
  for (int t = 0; t < NT; ++t) {
    const int n = nBase + 16 * t + (lane & 15);
#pragma unroll
    for (int i = 0; i < 8; ++i) {
      int m = mBase + i + ((lane < 16) ? 0 : 8);
      if (m < M && n < N) {
        float v = acc[t][i];
        if (bias) v += bias[n];
        if (act == ACT_LRELU) v = lrelu(v);
        C[(size_t)m * N + n] = v;
      }
    }
  }
}

// ---- WMMA implicit-GEMM 3x3 conv, pad 1, NCHW, fused lrelu + residual. -----
// The scattered im2col A-gather is staged once and amortized over NT N-tiles.
template <int NT>
__global__ void k_conv3_wmma(const float* __restrict__ in, const float* __restrict__ w,
                             const float* __restrict__ bias, const float* __restrict__ resid,
                             float* __restrict__ out, int Cin, int Cout, int H, int Wd, int act) {
  __shared__ unsigned short sA[16][32];
  __shared__ unsigned short sB[NT][16][32];
  const int lane = threadIdx.x;
  const int HW = H * Wd;
  const int M = 4 * HW;
  const int K = Cin * 9;
  const int mBase = blockIdx.x * 16;
  const int nBase = blockIdx.y * (16 * NT);
  v8f acc[NT];
#pragma unroll
  for (int t = 0; t < NT; ++t)
    for (int i = 0; i < 8; ++i) acc[t][i] = 0.f;

  for (int kk = 0; kk < K; kk += 32) {
    // im2col gather of the A tile (scattered; scalar by necessity)
#pragma unroll
    for (int j = 0; j < 16; ++j) {
      int idx = lane * 16 + j;
      int r = idx >> 5, kc = idx & 31;
      int m = mBase + r, k = kk + kc;
      float av = 0.f;
      if (m < M && k < K) {
        int img = m / HW, rem = m - img * HW;
        int oy = rem / Wd, ox = rem - oy * Wd;
        int ci = k / 9, rr = k - ci * 9;
        int ky = rr / 3, kx = rr - ky * 3;
        int iy = oy + ky - 1, ix = ox + kx - 1;
        if (iy >= 0 && iy < H && ix >= 0 && ix < Wd)
          av = in[(((size_t)img * Cin + ci) * H + iy) * Wd + ix];
      }
      sA[r][kc] = f2bf(av);
    }
#pragma unroll
    for (int t = 0; t < NT; ++t)
      stage_tile(w, nBase + 16 * t, Cout, K, kk, K, sB[t], lane);
    __syncthreads();
    v16bf a = frag_a(sA, lane);
#pragma unroll
    for (int t = 0; t < NT; ++t) {
      v16bf b = frag_b(sB[t], lane);
      acc[t] = __builtin_amdgcn_wmma_f32_16x16x32_bf16(false, a, false, b,
                                                       (short)0, acc[t], false, false);
    }
    __syncthreads();
  }
#pragma unroll
  for (int t = 0; t < NT; ++t) {
    const int n = nBase + 16 * t + (lane & 15);
#pragma unroll
    for (int i = 0; i < 8; ++i) {
      int m = mBase + i + ((lane < 16) ? 0 : 8);
      if (m < M && n < Cout) {
        float v = acc[t][i] + bias[n];
        if (act == ACT_LRELU) v = lrelu(v);
        int img = m / HW, rem = m - img * HW;
        size_t oidx = (((size_t)img * Cout + n) * HW) + rem;
        if (resid) v += resid[oidx];
        out[oidx] = v;
      }
    }
  }
}

// ------------------------------- small direct conv (tiny Cin inputs) --------
__global__ void k_conv3_direct(const float* __restrict__ in, const float* __restrict__ w,
                               const float* __restrict__ bias, float* __restrict__ out,
                               int Cin, int Cout, int H, int Wd) {
  int idx = blockIdx.x * blockDim.x + threadIdx.x;
  int total = 4 * Cout * H * Wd;
  if (idx >= total) return;
  int x = idx % Wd; int t = idx / Wd;
  int y = t % H; t /= H;
  int co = t % Cout; int img = t / Cout;
  float acc = bias[co];
  for (int ci = 0; ci < Cin; ++ci)
    for (int ky = 0; ky < 3; ++ky) {
      int iy = y + ky - 1; if (iy < 0 || iy >= H) continue;
      for (int kx = 0; kx < 3; ++kx) {
        int ix = x + kx - 1; if (ix < 0 || ix >= Wd) continue;
        acc += in[(((size_t)img * Cin + ci) * H + iy) * Wd + ix] *
               w[(((size_t)co * Cin + ci) * 3 + ky) * 3 + kx];
      }
    }
  out[idx] = acc;
}

// ------------------------------- bilinear 4x upsample (half-pixel) ----------
__global__ void k_bilinear4x(const float* __restrict__ in, float* __restrict__ out) {
  int idx = blockIdx.x * blockDim.x + threadIdx.x;
  if (idx >= 65536) return;                 // (4,4,64,64)
  int x = idx & 63, y = (idx >> 6) & 63, c = (idx >> 12) & 3, img = idx >> 14;
  float sy = (y + 0.5f) * 0.25f - 0.5f;
  float sx = (x + 0.5f) * 0.25f - 0.5f;
  int y0 = (int)floorf(sy); float fy = sy - (float)y0;
  int x0 = (int)floorf(sx); float fx = sx - (float)x0;
  int y0c = iclamp(y0, 0, 15), y1c = iclamp(y0 + 1, 0, 15);
  int x0c = iclamp(x0, 0, 15), x1c = iclamp(x0 + 1, 0, 15);
  const float* p = in + ((size_t)img * 4 + c) * 256;
  float v = (1.f - fy) * ((1.f - fx) * p[y0c * 16 + x0c] + fx * p[y0c * 16 + x1c]) +
            fy        * ((1.f - fx) * p[y1c * 16 + x0c] + fx * p[y1c * 16 + x1c]);
  out[idx] = v;
}

// ------------------------------- contourlet decompose / reconstruct ---------
__global__ void k_ct_dec(const float* __restrict__ f, float* __restrict__ low,
                         float* __restrict__ sub) {
  int idx = blockIdx.x * blockDim.x + threadIdx.x;
  if (idx >= 131072) return;               // (4,32,32,32)
  int x = idx & 31, y = (idx >> 5) & 31, c = (idx >> 10) & 31, img = idx >> 15;
  const float* p = f + (((size_t)img * 32 + c) * 64 + 2 * y) * 64 + 2 * x;
  float a = p[0], b = p[1], cc = p[64], d = p[65];
  low[idx] = 0.5f * (a + b + cc + d);
  float s[8] = {0.5f * (a - b + cc - d), 0.5f * (a + b - cc - d), 0.5f * (a - b - cc + d),
                a - b, cc - d, a - cc, b - d, a - d};
  for (int j = 0; j < 8; ++j)
    sub[((((size_t)img * 256 + j * 32 + c) * 32) + y) * 32 + x] = s[j];
}

__global__ void k_ct_rec(const float* __restrict__ low, const float* __restrict__ sub,
                         float* __restrict__ out) {
  int idx = blockIdx.x * blockDim.x + threadIdx.x;
  if (idx >= 131072) return;
  int x = idx & 31, y = (idx >> 5) & 31, c = (idx >> 10) & 31, img = idx >> 15;
  float g[8];
  for (int j = 0; j < 8; ++j)
    g[j] = sub[((((size_t)img * 256 + j * 32 + c) * 32) + y) * 32 + x];
  float lo = low[idx];
  float h1 = g[0], h2 = g[1], h3 = g[2], e1 = g[3], e2 = g[4], e3 = g[5], e4 = g[6], e5 = g[7];
  float a = 0.5f * (lo + h1 + h2 + h3) + 0.05f * (e1 + e3 + e5);
  float b = 0.5f * (lo - h1 + h2 - h3) + 0.05f * (-e1 + e4);
  float cv = 0.5f * (lo + h1 - h2 - h3) + 0.05f * (e2 - e3);
  float d = 0.5f * (lo - h1 - h2 + h3) + 0.05f * (-e2 - e4 - e5);
  float* q = out + (((size_t)img * 32 + c) * 64 + 2 * y) * 64 + 2 * x;
  q[0] = a; q[1] = b; q[64] = cv; q[65] = d;
}

// ------------------------------- batchnorm ----------------------------------
__global__ void k_bnstats(const float* __restrict__ r, float* __restrict__ stats,
                          int C, int HW) {
  int c = blockIdx.x, t = threadIdx.x;
  __shared__ float ssum[256], ssq[256];
  float s = 0.f, q = 0.f;
  int total = 4 * HW;
  for (int i = t; i < total; i += 256) {
    int img = i / HW, p = i - img * HW;
    float v = r[(((size_t)img * C + c) * HW) + p];
    s += v; q += v * v;
  }
  ssum[t] = s; ssq[t] = q; __syncthreads();
  for (int o = 128; o > 0; o >>= 1) {
    if (t < o) { ssum[t] += ssum[t + o]; ssq[t] += ssq[t + o]; }
    __syncthreads();
  }
  if (t == 0) {
    float mean = ssum[0] / (float)total;
    stats[c * 2] = mean;
    stats[c * 2 + 1] = ssq[0] / (float)total - mean * mean;
  }
}

__global__ void k_bnapply_add(float* __restrict__ x, const float* __restrict__ r,
                              const float* __restrict__ stats, const float* __restrict__ g,
                              const float* __restrict__ b, int C, int HW, int total) {
  int idx = blockIdx.x * blockDim.x + threadIdx.x;
  if (idx >= total) return;
  int c = (idx / HW) % C;
  float mean = stats[c * 2], var = stats[c * 2 + 1];
  float xh = (r[idx] - mean) * rsqrtf(var + 1e-5f);
  x[idx] += xh * g[c] + b[c];
}

// ------------------------------- layernorm (D=32) ---------------------------
__global__ void k_layernorm(const float* __restrict__ in, const float* __restrict__ g,
                            const float* __restrict__ b, float* __restrict__ out, int M) {
  int m = blockIdx.x * blockDim.x + threadIdx.x;
  if (m >= M) return;
  const float* p = in + (size_t)m * 32;
  float s = 0.f;
  for (int j = 0; j < 32; ++j) s += p[j];
  float mu = s * (1.f / 32.f);
  float q = 0.f;
  for (int j = 0; j < 32; ++j) { float d = p[j] - mu; q += d * d; }
  float rs = rsqrtf(q * (1.f / 32.f) + 1e-5f);
  for (int j = 0; j < 32; ++j) out[(size_t)m * 32 + j] = (p[j] - mu) * rs * g[j] + b[j];
}

// --------- mamba depthwise causal conv1d (k=4, pad 3 left) + bias + silu ----
__global__ void k_dwconv_silu(const float* __restrict__ xz, const float* __restrict__ w,
                              const float* __restrict__ bias, float* __restrict__ out) {
  int idx = blockIdx.x * blockDim.x + threadIdx.x;
  if (idx >= 4 * 4096 * 64) return;
  int c = idx & 63, l = (idx >> 6) & 4095, bimg = idx >> 18;
  float acc = bias[c];
  for (int t = 0; t < 4; ++t) {
    int li = l - 3 + t;
    if (li >= 0) acc += w[c * 4 + t] * xz[(((size_t)bimg * 4096 + li) * 128) + c];
  }
  out[idx] = siluf(acc);
}

// --------- dt = softplus(xdbl[:, :2] @ dt_w^T + dt_b) -----------------------
__global__ void k_dt(const float* __restrict__ xdbl, const float* __restrict__ dtw,
                     const float* __restrict__ dtb, float* __restrict__ dt) {
  int idx = blockIdx.x * blockDim.x + threadIdx.x;
  if (idx >= 16384 * 64) return;
  int d = idx & 63, m = idx >> 6;
  float v = xdbl[(size_t)m * 34] * dtw[d * 2] + xdbl[(size_t)m * 34 + 1] * dtw[d * 2 + 1] + dtb[d];
  dt[idx] = v > 20.f ? v : log1pf(expf(v));
}

// --------- selective scan + D skip + silu(z) gate, per (b,d) thread ---------
__global__ void k_scan(const float* __restrict__ Alog, const float* __restrict__ Dv,
                       const float* __restrict__ dt, const float* __restrict__ xm_s,
                       const float* __restrict__ xdbl, const float* __restrict__ xz,
                       float* __restrict__ y) {
  int gid = blockIdx.x * blockDim.x + threadIdx.x;
  if (gid >= 256) return;
  int b = gid >> 6, d = gid & 63;
  float A[16], h[16];
  for (int s = 0; s < 16; ++s) { A[s] = -expf(Alog[d * 16 + s]); h[s] = 0.f; }
  float Dd = Dv[d];
  for (int l = 0; l < 4096; ++l) {
    size_t base = (size_t)b * 4096 + l;
    float dtv = dt[base * 64 + d];
    float xv  = xm_s[base * 64 + d];
    const float* xd = xdbl + base * 34;
    float acc = 0.f;
#pragma unroll
    for (int s = 0; s < 16; ++s) {
      float dA = __expf(dtv * A[s]);
      h[s] = dA * h[s] + dtv * xd[2 + s] * xv;
      acc += h[s] * xd[18 + s];
    }
    float yv = acc + xv * Dd;
    float zv = xz[base * 128 + 64 + d];
    y[base * 64 + d] = yv * siluf(zv);
  }
}

// ------------------------------- elementwise --------------------------------
__global__ void k_addinto(float* __restrict__ dst, const float* __restrict__ a, int n) {
  int i = blockIdx.x * blockDim.x + threadIdx.x;
  if (i < n) dst[i] += a[i];
}
__global__ void k_add2(float* __restrict__ dst, const float* __restrict__ a,
                       const float* __restrict__ b, int n) {
  int i = blockIdx.x * blockDim.x + threadIdx.x;
  if (i < n) dst[i] = a[i] + b[i];
}
__global__ void k_zero(float* __restrict__ dst, int n) {
  int i = blockIdx.x * blockDim.x + threadIdx.x;
  if (i < n) dst[i] = 0.f;
}
__global__ void k_nchw_tokens(const float* __restrict__ in, float* __restrict__ out,
                              int C, int HW) {
  int idx = blockIdx.x * blockDim.x + threadIdx.x;
  int total = 4 * C * HW;
  if (idx >= total) return;
  int p = idx % HW; int t = idx / HW;
  int c = t % C; int img = t / C;
  out[((size_t)img * HW + p) * C + c] = in[idx];
}

// ------------------------------- final linears ------------------------------
__global__ void k_lin1(const float* __restrict__ s, const float* __restrict__ w,
                       const float* __restrict__ b, float* __restrict__ out) {
  int n = blockIdx.x, bb = blockIdx.y, t = threadIdx.x;
  __shared__ float red[256];
  float acc = 0.f;
  const float* sp = s + (size_t)bb * 131072;
  const float* wp = w + (size_t)n * 131072;
  for (int k = t; k < 131072; k += 256) acc += sp[k] * wp[k];
  red[t] = acc; __syncthreads();
  for (int o = 128; o > 0; o >>= 1) { if (t < o) red[t] += red[t + o]; __syncthreads(); }
  if (t == 0) out[bb * 32 + n] = red[0] + b[n];
}
__global__ void k_lin2(const float* __restrict__ t1, const float* __restrict__ w,
                       const float* __restrict__ b, float* __restrict__ out) {
  int idx = threadIdx.x;
  if (idx >= 40) return;
  int bb = idx / 10, j = idx % 10;
  float acc = b[j];
  for (int n = 0; n < 32; ++n) acc += t1[bb * 32 + n] * w[j * 32 + n];
  out[idx] = acc;
}

// ---------------------------------------------------------------------------
// Host orchestration.  Param leaf indices assume JAX pytree order (dict keys
// sorted) of setup_inputs(): ms, pan, then params{...}.
// ---------------------------------------------------------------------------
enum {
  IDX_MS = 0, IDX_PAN = 1,
  IDX_INV = 2,        // 8 mamba blocks x 11 leaves
  IDX_LINN = 90,      // 2 rev blocks x 4
  IDX_LIN1 = 98, IDX_LIN2 = 100,
  IDX_MENC = 102,     // 3 hin x 8
  IDX_MIN = 126,
  IDX_MMAMBA = 128,   // 8 x 11
  IDX_MPART = 216,
  IDX_PENC = 218,     // 3 x 8
  IDX_PIN = 242,
  IDX_PMAMBA = 244,   // 8 x 11
  IDX_PPART = 332,
  IDX_SINN = 334      // 2 x 4  -> 342 total
};
// mamba leaf offsets (sorted keys): A_log,D,conv_b,conv_w,dt_b,dt_w,in_proj,ln_b,ln_g,out_proj,x_proj
// hin leaf offsets: bn1.b,bn1.g,bn2.b,bn2.g,c1.b,c1.w,c2.b,c2.w
// rev leaf offsets: f.b,f.w,g.b,g.w

extern "C" void kernel_launch(void* const* d_in, const int* in_sizes, int n_in,
                              void* d_out, int out_size, void* d_ws, size_t ws_size,
                              hipStream_t stream) {
  (void)in_sizes; (void)n_in; (void)out_size; (void)ws_size;
  auto IN = [&](int i) -> const float* { return (const float*)d_in[i]; };

  float* ws = (float*)d_ws;
  size_t off = 0;
  auto alloc = [&](size_t n) { float* p = ws + off; off += n; return p; };

  float* m_up  = alloc(65536);
  float* mfeat = alloc(524288);
  float* pfeat = alloc(524288);
  float* l_x1  = alloc(131072);   // m_l
  float* l_x2  = alloc(131072);   // p_l
  float* s_x1  = alloc(1048576);  // m_s
  float* s_x2  = alloc(1048576);  // p_s
  float* m_img = alloc(524288);
  float* p_img = alloc(524288);
  float* r_tmp = alloc(524288);
  float* stats = alloc(64);
  float* tokA  = alloc(524288);
  float* tokB  = alloc(524288);
  float* tokC  = alloc(524288);
  float* tokD  = alloc(524288);
  float* tokE  = alloc(524288);
  float* tokF  = alloc(524288);
  float* lnbuf = alloc(524288);
  float* xz    = alloc(2097152);
  float* xm_s  = alloc(1048576);
  float* xdbl  = alloc(557056);
  float* dtbuf = alloc(1048576);
  float* ybuf  = alloc(1048576);
  float* sumb  = alloc(524288);
  float* lin1o = alloc(128);

  auto gemm = [&](const float* A, const float* Wt, const float* bias, float* C,
                  int M, int K, int N, int act) {
    dim3 g((M + 15) / 16, (N + 31) / 32);
    k_gemm_wmma<2><<<g, dim3(32), 0, stream>>>(A, Wt, bias, C, M, K, N, act);
  };
  auto conv3 = [&](const float* in, const float* w, const float* bias, const float* resid,
                   float* out, int Cin, int Cout, int H, int act) {
    int M = 4 * H * H;
    if (Cout >= 64) {
      dim3 g((M + 15) / 16, (Cout + 63) / 64);
      k_conv3_wmma<4><<<g, dim3(32), 0, stream>>>(in, w, bias, resid, out, Cin, Cout, H, H, act);
    } else {
      dim3 g((M + 15) / 16, (Cout + 31) / 32);
      k_conv3_wmma<2><<<g, dim3(32), 0, stream>>>(in, w, bias, resid, out, Cin, Cout, H, H, act);
    }
  };
  auto ew = [&](int n) { return dim3((n + 255) / 256); };

  // ---- front end -----------------------------------------------------------
  k_bilinear4x<<<ew(65536), 256, 0, stream>>>(IN(IDX_MS), m_up);
  k_conv3_direct<<<ew(524288), 256, 0, stream>>>(m_up, IN(IDX_MIN + 1), IN(IDX_MIN), mfeat, 4, 32, 64, 64);
  k_conv3_direct<<<ew(524288), 256, 0, stream>>>(IN(IDX_PAN), IN(IDX_PIN + 1), IN(IDX_PIN), pfeat, 1, 32, 64, 64);

  k_ct_dec<<<ew(131072), 256, 0, stream>>>(mfeat, l_x1, s_x1);
  k_ct_dec<<<ew(131072), 256, 0, stream>>>(pfeat, l_x2, s_x2);

  // ---- revnets (WMMA implicit-GEMM, fused lrelu + residual add) ------------
  for (int blk = 0; blk < 2; ++blk) {
    int pb = IDX_SINN + blk * 4;
    conv3(s_x2, IN(pb + 1), IN(pb + 0), s_x1, s_x1, 256, 256, 32, ACT_LRELU);
    conv3(s_x1, IN(pb + 3), IN(pb + 2), s_x2, s_x2, 256, 256, 32, ACT_LRELU);
  }
  for (int blk = 0; blk < 2; ++blk) {
    int pb = IDX_LINN + blk * 4;
    conv3(l_x2, IN(pb + 1), IN(pb + 0), l_x1, l_x1, 32, 32, 32, ACT_LRELU);
    conv3(l_x1, IN(pb + 3), IN(pb + 2), l_x2, l_x2, 32, 32, 32, ACT_LRELU);
  }

  k_ct_rec<<<ew(131072), 256, 0, stream>>>(l_x1, s_x2, m_img);  // m_out = rec(m_outl, p_outs)
  k_ct_rec<<<ew(131072), 256, 0, stream>>>(l_x2, s_x1, p_img);  // p_out = rec(p_outl, m_outs)

  // ---- HIN encoder blocks --------------------------------------------------
  auto hin = [&](float* x, int pbase) {
    conv3(x, IN(pbase + 5), IN(pbase + 4), nullptr, r_tmp, 32, 32, 64, ACT_LRELU);
    k_bnstats<<<32, 256, 0, stream>>>(r_tmp, stats, 32, 4096);
    k_bnapply_add<<<ew(524288), 256, 0, stream>>>(x, r_tmp, stats, IN(pbase + 1), IN(pbase + 0), 32, 4096, 524288);
    conv3(x, IN(pbase + 7), IN(pbase + 6), nullptr, r_tmp, 32, 32, 64, ACT_LRELU);
    k_bnstats<<<32, 256, 0, stream>>>(r_tmp, stats, 32, 4096);
    k_bnapply_add<<<ew(524288), 256, 0, stream>>>(x, r_tmp, stats, IN(pbase + 3), IN(pbase + 2), 32, 4096, 524288);
  };
  for (int i = 0; i < 3; ++i) hin(m_img, IDX_MENC + 8 * i);
  for (int i = 0; i < 3; ++i) hin(p_img, IDX_PENC + 8 * i);

  // ---- 1x1 partition convs, emitted directly in token layout [b,l,c] -------
  k_nchw_tokens<<<ew(524288), 256, 0, stream>>>(m_img, lnbuf, 32, 4096);
  gemm(lnbuf, IN(IDX_MPART + 1), IN(IDX_MPART), tokA, 16384, 32, 32, ACT_NONE);
  k_nchw_tokens<<<ew(524288), 256, 0, stream>>>(p_img, lnbuf, 32, 4096);
  gemm(lnbuf, IN(IDX_PPART + 1), IN(IDX_PPART), tokB, 16384, 32, 32, ACT_NONE);

  // ---- mamba block ---------------------------------------------------------
  auto mamba = [&](int pb, const float* tin, float* tout) {
    const float* Alog = IN(pb + 0); const float* Dv = IN(pb + 1);
    const float* cb = IN(pb + 2);   const float* cw = IN(pb + 3);
    const float* dtb = IN(pb + 4);  const float* dtw = IN(pb + 5);
    const float* inp = IN(pb + 6);  const float* lnb = IN(pb + 7);
    const float* lng = IN(pb + 8);  const float* outp = IN(pb + 9);
    const float* xp = IN(pb + 10);
    k_layernorm<<<ew(16384), 256, 0, stream>>>(tin, lng, lnb, lnbuf, 16384);
    gemm(lnbuf, inp, nullptr, xz, 16384, 32, 128, ACT_NONE);
    k_dwconv_silu<<<ew(1048576), 256, 0, stream>>>(xz, cw, cb, xm_s);
    gemm(xm_s, xp, nullptr, xdbl, 16384, 64, 34, ACT_NONE);
    k_dt<<<ew(1048576), 256, 0, stream>>>(xdbl, dtw, dtb, dtbuf);
    k_scan<<<dim3(8), dim3(32), 0, stream>>>(Alog, Dv, dtbuf, xm_s, xdbl, xz, ybuf);
    gemm(ybuf, outp, nullptr, tout, 16384, 64, 32, ACT_NONE);
  };

  // ---- inv_mamba: x=m_t, res=p_t; res+=x; y=mamba(res); x,res = res,y ------
  const int NT = 524288;
  float* x = tokA; float* res = tokB; float* spare = tokC;
  for (int i = 0; i < 8; ++i) {
    k_addinto<<<ew(NT), 256, 0, stream>>>(res, x, NT);
    mamba(IDX_INV + 11 * i, res, spare);
    float* nx = res; float* nr = spare; spare = x; x = nx; res = nr;
  }
  float* m_t = x; float* p_t = res;  // spare is free

  // ---- m_mamba residual chain ---------------------------------------------
  float* resm = tokD; float* outm = tokE;
  k_zero<<<ew(NT), 256, 0, stream>>>(resm, NT);
  float* cur = m_t;
  for (int i = 0; i < 8; ++i) {
    k_addinto<<<ew(NT), 256, 0, stream>>>(resm, cur, NT);
    mamba(IDX_MMAMBA + 11 * i, resm, outm);
    float* t = cur; cur = outm; outm = (t == p_t) ? spare : t;  // never clobber p_t
    if (outm == p_t) outm = spare;
  }
  m_t = cur;

  // ---- p_mamba residual chain ---------------------------------------------
  float* resp = tokF; float* outp2 = (m_t == tokE) ? tokD : tokE;
  if (outp2 == m_t) outp2 = spare;
  k_zero<<<ew(NT), 256, 0, stream>>>(resp, NT);
  cur = p_t;
  for (int i = 0; i < 8; ++i) {
    k_addinto<<<ew(NT), 256, 0, stream>>>(resp, cur, NT);
    mamba(IDX_PMAMBA + 11 * i, resp, outp2);
    float* t = cur; cur = outp2; outp2 = (t == m_t) ? spare : t;
    if (outp2 == m_t) outp2 = spare;
  }
  p_t = cur;

  // ---- head ----------------------------------------------------------------
  k_add2<<<ew(NT), 256, 0, stream>>>(sumb, m_t, p_t, NT);
  k_lin1<<<dim3(32, 4), dim3(256), 0, stream>>>(sumb, IN(IDX_LIN1 + 1), IN(IDX_LIN1), lin1o);
  k_lin2<<<dim3(1), dim3(64), 0, stream>>>(lin1o, IN(IDX_LIN2 + 1), IN(IDX_LIN2), (float*)d_out);
}
